// RetinanetFPNResNets_4629974745421
// MI455X (gfx1250) — matrix operand order repaired and checked by reading, not compiled, not run on hardware
//
#include <hip/hip_runtime.h>
#include <hip/hip_bf16.h>
#include <math.h>

typedef __attribute__((ext_vector_type(16))) _Float16 v16h;
typedef __attribute__((ext_vector_type(8)))  float    v8f;

// ---------------- constants ----------------
constexpr int kNA   = 76725;              // total anchors
constexpr int kG    = 20;                 // max GT boxes
constexpr int kB    = 4;                  // batch
constexpr int kKDIM = 2304;               // 9*256
constexpr int kCH   = 72;                 // K chunks of 32 (8 chunks per 3x3 tap)
// d_out layout: anchors[306900] | cls[4*76725*80] | reg[4*76725*4] | loss_cls | loss_reg
constexpr int kAnchorsN = kNA * 4;
constexpr int kClsN     = kB * kNA * 80;
constexpr int kRegN     = kB * kNA * 4;

// ---------------- small helpers ----------------
__device__ inline float blockReduce256(float v) {
    __shared__ float s[256];
    s[threadIdx.x] = v; __syncthreads();
    for (int off = 128; off > 0; off >>= 1) {
        if ((int)threadIdx.x < off) s[threadIdx.x] += s[threadIdx.x + off];
        __syncthreads();
    }
    return s[0];
}
__device__ inline unsigned int ordbits(float x) {
    unsigned int u = __float_as_uint(x);
    return (u & 0x80000000u) ? ~u : (u | 0x80000000u);
}

// ---------------- init scratch ----------------
__global__ void init_kernel(unsigned long long* keys, float* acc) {
    int t = threadIdx.x;
    if (t < kB * kG) keys[t] = 0ull;
    if (t < 4) acc[t] = 0.0f;
}

// ---------------- NCHW f32 -> NHWC f16 tiled transpose ----------------
__global__ __launch_bounds__(256) void nchw_to_nhwc_f16_kernel(
        const float* __restrict__ src, _Float16* __restrict__ dst, int HW) {
    int b = blockIdx.z;
    int cinBase = blockIdx.y * 32;
    int pBase = blockIdx.x * 32;
    __shared__ _Float16 T[32][33];
    int px = threadIdx.x & 31;
    int cy = threadIdx.x >> 5;
#pragma unroll
    for (int i = 0; i < 4; ++i) {
        int cl = cy + i * 8;
        int p = pBase + px;
        float v = 0.0f;
        if (p < HW) v = src[((size_t)b * 256 + cinBase + cl) * HW + p];
        T[cl][px] = (_Float16)v;
    }
    __syncthreads();
    int cx = threadIdx.x & 31;
    int py = threadIdx.x >> 5;
#pragma unroll
    for (int i = 0; i < 4; ++i) {
        int pl = py + i * 8;
        int p = pBase + pl;
        if (p < HW)
            dst[((size_t)b * HW + p) * 256 + cinBase + cx] = T[cx][pl];
    }
}

// ---------------- pack weights into A-fragment layout ----------------
// GEMM K order: k = r*256 + cin (r = ky*3+kx). chunk c: r=c>>3, cinBase=(c&7)*32
__global__ void pack_weights_kernel(const float* __restrict__ w,
                                    _Float16* __restrict__ dst,
                                    int MT, int Cout, int total) {
    int idx = blockIdx.x * 256 + threadIdx.x;   // (tile, c, lane)
    if (idx >= total) return;
    int lane = idx & 31;
    int c    = (idx >> 5) % kCH;
    int tile = idx / (kCH * 32);
    int m = tile * 16 + (lane & 15);
    int h = lane >> 4;
    _Float16 vals[16];
#pragma unroll
    for (int j = 0; j < 16; ++j) {
        int v = j >> 1, tb = j & 1;
        int base = (v < 4) ? (2 * v) : (16 + 2 * (v - 4));
        int kl = base + 8 * h + tb;
        int kk = c * 32 + kl;             // GEMM k
        int r  = kk >> 8;                 // 0..8
        int ci = kk & 255;
        float f = (m < Cout) ? w[(size_t)m * kKDIM + ci * 9 + r] : 0.0f;
        vals[j] = (_Float16)f;
    }
    *(v16h*)&dst[(size_t)idx * 16] = *(v16h*)vals;
}

// ---------------- WMMA implicit-GEMM conv3x3 SAME, Cin=256, NHWC ----------------
// NT*16 positions per block, 8 waves, TPW M-tiles per wave.
// MODE 0: +bias, ReLU, vectorized f16 NHWC store
// MODE 1: +bias, reg preds f32
// MODE 2: +bias, sigmoid, cls preds f32
template <int TPW, int NT, int MODE>
__global__ __launch_bounds__(256) void conv_wmma_kernel(
        const _Float16* __restrict__ in,      // [B][HW][256] f16
        const _Float16* __restrict__ wpack,
        const float* __restrict__ bias,
        _Float16* __restrict__ outAct,        // MODE 0, NHWC
        float* __restrict__ outPred,          // MODE 1/2
        int H, int W, int MT, int Cout, int anchorOff) {
    const int HW    = H * W;
    const int b     = blockIdx.y;
    const int tid   = threadIdx.x;
    const int lane  = tid & 31;
    const int wave  = tid >> 5;
    const int pbase = blockIdx.x * (NT * 16);

    const _Float16* inB = in + (size_t)b * HW * 256;

    __shared__ alignas(16) _Float16 Bs[2][NT][512];

    // staging: threads 0..NT*64-1, each loads 16B (8 channels) for one position
    const int n_all = tid >> 2;                  // position within block tile
    const int q     = tid & 3;                   // which 8-channel group
    const bool stg  = n_all < NT * 16;
    const int ps    = pbase + n_all;
    const bool pv   = stg && (ps < HW);
    const int yy    = pv ? (ps / W) : 0;
    const int xx    = pv ? (ps - yy * W) : 0;
    const int nt_s  = n_all >> 4;                // which N-tile
    const int row_s = (n_all & 15) + ((q & 1) ? 16 : 0);
    const int off_s = (q >> 1) * 8;

    v8f acc[TPW][NT] = {};

    for (int c = 0; c < kCH; ++c) {
        const int r  = c >> 3;                   // tap index, wave-uniform
        const int ky = r / 3, kx = r - ky * 3;
        const int cinBase = (c & 7) * 32;
        if (stg) {
            uint4 data = make_uint4(0u, 0u, 0u, 0u);
            int iy = yy + ky - 1, ix = xx + kx - 1;
            if (pv && iy >= 0 && iy < H && ix >= 0 && ix < W)
                data = *(const uint4*)&inB[((size_t)(iy * W + ix)) * 256 + cinBase + q * 8];
            *(uint4*)&Bs[c & 1][nt_s][row_s * 16 + off_s] = data;
        }
        __syncthreads();
        // prefetch next chunk's A-fragment stream (global_prefetch_b8)
        if (c + 1 < kCH)
            __builtin_prefetch(
                &wpack[(((size_t)(wave * TPW) * kCH + (c + 1)) * 32 + lane) * 16], 0, 0);
        v16h bf[NT];
#pragma unroll
        for (int nt = 0; nt < NT; ++nt)
            bf[nt] = *(const v16h*)&Bs[c & 1][nt][lane * 16];
#pragma unroll
        for (int t = 0; t < TPW; ++t) {
            int mt = wave * TPW + t;             // wave-uniform
            if (mt < MT) {
                v16h af = *(const v16h*)&wpack[(((size_t)mt * kCH + c) * 32 + lane) * 16];
#pragma unroll
                for (int nt = 0; nt < NT; ++nt)
                    acc[t][nt] = __builtin_amdgcn_wmma_f32_16x16x32_f16(
                        false, af, false, bf[nt], (short)0, acc[t][nt], false, false);
            }
        }
    }

    // epilogue: D layout — lane l, row r: co_local = r + 8*(l>>4), n = l&15
    const int h  = lane >> 4;
    const int nn = lane & 15;
#pragma unroll
    for (int nt = 0; nt < NT; ++nt) {
        int p = pbase + nt * 16 + nn;
        if (p >= HW) continue;
#pragma unroll
        for (int t = 0; t < TPW; ++t) {
            int mt = wave * TPW + t;
            if (mt >= MT) continue;
            if constexpr (MODE == 0) {
                int co0 = mt * 16 + 8 * h;       // 8 consecutive channels
                _Float16 vals[8];
#pragma unroll
                for (int rr = 0; rr < 8; ++rr) {
                    float v = acc[t][nt][rr] + bias[co0 + rr];
                    vals[rr] = (_Float16)fmaxf(v, 0.0f);
                }
                *(uint4*)&(outAct + (size_t)b * HW * 256)[(size_t)p * 256 + co0] =
                    *(uint4*)vals;
            } else if constexpr (MODE == 1) {
#pragma unroll
                for (int rr = 0; rr < 8; ++rr) {
                    int co = mt * 16 + rr + 8 * h;
                    if (co < Cout) {
                        float v = acc[t][nt][rr] + bias[co];
                        int ai = co >> 2, j = co & 3;
                        int ag = anchorOff + p * 9 + ai;
                        outPred[((size_t)b * kNA + ag) * 4 + j] = v;
                    }
                }
            } else {
#pragma unroll
                for (int rr = 0; rr < 8; ++rr) {
                    int co = mt * 16 + rr + 8 * h;
                    if (co < Cout) {
                        float v = acc[t][nt][rr] + bias[co];
                        int ai = co / 80, cl = co - ai * 80;
                        int ag = anchorOff + p * 9 + ai;
                        outPred[((size_t)b * kNA + ag) * 80 + cl] =
                            1.0f / (1.0f + expf(-v));
                    }
                }
            }
        }
    }
}

// ---------------- anchors ----------------
__global__ void anchors_kernel(float* __restrict__ anch) {
    int i = blockIdx.x * 256 + threadIdx.x;
    if (i >= kNA) return;
    const int Ws[5]    = {80, 40, 20, 10, 5};
    const int AOFF[6]  = {0, 57600, 72000, 75600, 76500, 76725};
    const float SIZ[5] = {32.f, 64.f, 128.f, 256.f, 512.f};
    const float STR[5] = {8.f, 16.f, 32.f, 64.f, 128.f};
    const float SCL[3] = {1.0f, 1.2599210498948732f, 1.5874010519681994f};
    const float RAT[3] = {0.5f, 1.0f, 2.0f};
    int lev = 4;
    for (int l = 0; l < 4; ++l) if (i < AOFF[l + 1]) { lev = l; break; }
    int q = i - AOFF[lev];
    int cell = q / 9, a = q - cell * 9;
    int W = Ws[lev];
    int y = cell / W, x = cell - y * W;
    int ri = a / 3, si = a - ri * 3;
    float size = SIZ[lev];
    float ctr = (size - 1.0f) * 0.5f;
    float wsz = sqrtf(size * size / RAT[ri]) * SCL[si];
    float hsz = wsz * RAT[ri];
    float sx = x * STR[lev], sy = y * STR[lev];
    anch[i * 4 + 0] = sx + ctr - 0.5f * (wsz - 1.0f);
    anch[i * 4 + 1] = sy + ctr - 0.5f * (hsz - 1.0f);
    anch[i * 4 + 2] = sx + ctr + 0.5f * (wsz - 1.0f);
    anch[i * 4 + 3] = sy + ctr + 0.5f * (hsz - 1.0f);
}

// ---------------- targets pass 1 ----------------
__global__ void targets_iou_kernel(const float* __restrict__ anch,
                                   const float* __restrict__ gt,
                                   const int* __restrict__ ngt,
                                   const float* __restrict__ img,
                                   float* __restrict__ max_ov,
                                   int* __restrict__ argmax,
                                   unsigned long long* __restrict__ gtkeys) {
    int b = blockIdx.y;
    int a = blockIdx.x * 256 + threadIdx.x;
    __shared__ unsigned long long skey[kG];
    if ((int)threadIdx.x < kG) skey[threadIdx.x] = 0ull;
    __syncthreads();
    bool act = a < kNA;
    float a0 = 0, a1 = 0, a2 = 0, a3 = 0, areaA = 1;
    bool inside = false;
    if (act) {
        a0 = anch[a * 4 + 0]; a1 = anch[a * 4 + 1];
        a2 = anch[a * 4 + 2]; a3 = anch[a * 4 + 3];
        inside = (a0 >= 0.f) && (a1 >= 0.f) && (a2 < img[1]) && (a3 < img[0]);
        areaA = (a2 - a0 + 1.f) * (a3 - a1 + 1.f);
    }
    int n = ngt[b];
    float best = -2.0f; int bi = 0;
    if (act) {
        for (int i = 0; i < kG; ++i) {
            const float* g = &gt[((size_t)b * kG + i) * 5];
            float g0 = g[0], g1 = g[1], g2 = g[2], g3 = g[3];
            float areaB = (g2 - g0 + 1.f) * (g3 - g1 + 1.f);
            float iw = fminf(a2, g2) - fmaxf(a0, g0) + 1.f;
            float ih = fminf(a3, g3) - fmaxf(a1, g1) + 1.f;
            float inter = fmaxf(iw, 0.f) * fmaxf(ih, 0.f);
            float ov = inter / (areaA + areaB - inter);
            bool m = (i < n) && inside;
            ov = m ? ov : -1.0f;
            if (ov > best) { best = ov; bi = i; }
            unsigned long long key =
                ((unsigned long long)ordbits(ov) << 32) | (unsigned int)(~(unsigned int)a);
            atomicMax(&skey[i], key);
        }
        max_ov[(size_t)b * kNA + a] = best;
        argmax[(size_t)b * kNA + a] = bi;
    }
    __syncthreads();
    if ((int)threadIdx.x < kG)
        atomicMax(&gtkeys[b * kG + threadIdx.x], skey[threadIdx.x]);
}

// ---------------- targets pass 2 ----------------
__global__ void targets_override_kernel(const int* __restrict__ ngt,
                                        const unsigned long long* __restrict__ gtkeys,
                                        float* __restrict__ max_ov,
                                        int* __restrict__ argmax) {
    int b = threadIdx.x;
    if (b >= kB) return;
    int n = ngt[b];
    for (int i = 0; i < kG; ++i) {
        if (i < n) {
            unsigned long long k = gtkeys[b * kG + i];
            unsigned int a = ~(unsigned int)(k & 0xFFFFFFFFull);
            max_ov[(size_t)b * kNA + a] = 2.0f;
            argmax[(size_t)b * kNA + a] = i;
        }
    }
}

// ---------------- targets pass 3 ----------------
__global__ void targets_final_kernel(const float* __restrict__ anch,
                                     const float* __restrict__ gt,
                                     const float* __restrict__ img,
                                     const float* __restrict__ max_ov,
                                     const int* __restrict__ argmax,
                                     float* __restrict__ cls_t,
                                     float* __restrict__ reg_t,
                                     float* __restrict__ acc) {
    int b = blockIdx.y;
    int a = blockIdx.x * 256 + threadIdx.x;
    float pos = 0.0f;
    if (a < kNA) {
        float a0 = anch[a * 4 + 0], a1 = anch[a * 4 + 1];
        float a2 = anch[a * 4 + 2], a3 = anch[a * 4 + 3];
        bool inside = (a0 >= 0.f) && (a1 >= 0.f) && (a2 < img[1]) && (a3 < img[0]);
        float mo = max_ov[(size_t)b * kNA + a];
        int am = argmax[(size_t)b * kNA + a];
        const float* g = &gt[((size_t)b * kG + am) * 5];
        float cls = g[4];
        if (mo < 0.5f) cls = 0.0f;
        if (mo < 0.5f && mo > 0.4f) cls = -1.0f;
        float aw = a2 - a0 + 1.f, ah = a3 - a1 + 1.f;
        float ax = a0 + 0.5f * aw, ay = a1 + 0.5f * ah;
        float gw = g[2] - g[0] + 1.f, gh = g[3] - g[1] + 1.f;
        float gx = g[0] + 0.5f * gw, gy = g[1] + 0.5f * gh;
        float r0 = ((gx - ax) / aw) * 10.0f;
        float r1 = ((gy - ay) / ah) * 10.0f;
        float r2 = logf(gw / aw) * 5.0f;
        float r3 = logf(gh / ah) * 5.0f;
        if (!inside) { cls = -1.0f; r0 = r1 = r2 = r3 = 0.0f; }
        cls_t[(size_t)b * kNA + a] = cls;
        reg_t[((size_t)b * kNA + a) * 4 + 0] = r0;
        reg_t[((size_t)b * kNA + a) * 4 + 1] = r1;
        reg_t[((size_t)b * kNA + a) * 4 + 2] = r2;
        reg_t[((size_t)b * kNA + a) * 4 + 3] = r3;
        pos = (cls > 0.5f) ? 1.0f : 0.0f;
    }
    float s = blockReduce256(pos);
    if (threadIdx.x == 0) atomicAdd(&acc[0], s);
}

// ---------------- focal classification loss ----------------
__global__ void loss_cls_kernel(const float* __restrict__ probs,
                                const float* __restrict__ cls_t,
                                float* __restrict__ acc, int n) {
    int idx = blockIdx.x * 256 + threadIdx.x;
    float v = 0.0f;
    if (idx < n) {
        int cl = idx % 80;
        int ba = idx / 80;
        float ct = cls_t[ba];
        float valid = (ct > -0.5f) ? 1.0f : 0.0f;
        int ti = (int)ct;
        float t = (ct > 0.5f && (ti - 1) == cl) ? 1.0f : 0.0f;
        float p = probs[idx];
        float lp  = logf(fmaxf(p, 1e-12f));
        float lnp = logf(fmaxf(1.0f - p, 1e-12f));
        float ce = -(t * lp + (1.0f - t) * lnp);
        float pt = p * t + (1.0f - p) * (1.0f - t);
        float at = 0.25f * t + 0.75f * (1.0f - t);
        float om = 1.0f - pt;
        v = at * om * om * ce * valid;
    }
    float s = blockReduce256(v);
    if (threadIdx.x == 0) atomicAdd(&acc[1], s);
}

// ---------------- smooth-L1 regression loss ----------------
__global__ void loss_reg_kernel(const float* __restrict__ preds,
                                const float* __restrict__ cls_t,
                                const float* __restrict__ reg_t,
                                float* __restrict__ acc, int n) {
    constexpr float BETA = 1.0f / 9.0f;
    int idx = blockIdx.x * 256 + threadIdx.x;
    float v = 0.0f;
    if (idx < n) {
        int ba = idx >> 2;
        float pos = (cls_t[ba] > 0.5f) ? 1.0f : 0.0f;
        float d = fabsf(preds[idx] - reg_t[idx]);
        float sl1 = (d < BETA) ? (0.5f * d * d / BETA) : (d - 0.5f * BETA);
        v = sl1 * pos;
    }
    float s = blockReduce256(v);
    if (threadIdx.x == 0) atomicAdd(&acc[2], s);
}

__global__ void finalize_kernel(const float* __restrict__ acc, float* __restrict__ out) {
    float avg = fmaxf(acc[0], 1.0f);
    out[0] = acc[1] / avg;   // loss_cls
    out[1] = acc[2] / avg;   // loss_reg
}

// =============================================================================
extern "C" void kernel_launch(void* const* d_in, const int* in_sizes, int n_in,
                              void* d_out, int out_size, void* d_ws, size_t ws_size,
                              hipStream_t stream) {
    (void)in_sizes; (void)n_in; (void)out_size; (void)ws_size;
    const float* feat[5];
    for (int i = 0; i < 5; ++i) feat[i] = (const float*)d_in[i];
    const float *regW[5], *regB[5], *clsW[5], *clsB[5];
    for (int l = 0; l < 5; ++l) {
        regW[l] = (const float*)d_in[5 + l];
        regB[l] = (const float*)d_in[10 + l];
        clsW[l] = (const float*)d_in[15 + l];
        clsB[l] = (const float*)d_in[20 + l];
    }
    const float* gt  = (const float*)d_in[25];
    const int*   ngt = (const int*)d_in[26];
    const float* img = (const float*)d_in[27];

    // workspace layout
    char* ws = (char*)d_ws;
    size_t cur = 0;
    auto alloc = [&](size_t bytes) {
        size_t o = (cur + 255) & ~(size_t)255;
        cur = o + bytes;
        return o;
    };
    size_t pRegIn[4], pClsIn[4];
    const size_t innerPack = (size_t)16 * kCH * 512 * 2;
    for (int l = 0; l < 4; ++l) pRegIn[l] = alloc(innerPack);
    size_t pRegF = alloc((size_t)3 * kCH * 512 * 2);
    for (int l = 0; l < 4; ++l) pClsIn[l] = alloc(innerPack);
    size_t pClsF = alloc((size_t)45 * kCH * 512 * 2);
    const size_t actBytes = (size_t)kB * 256 * 6400 * 2;
    size_t oX = alloc(actBytes), oB1 = alloc(actBytes), oB2 = alloc(actBytes);
    size_t oMax  = alloc((size_t)kB * kNA * 4);
    size_t oArg  = alloc((size_t)kB * kNA * 4);
    size_t oClsT = alloc((size_t)kB * kNA * 4);
    size_t oRegT = alloc((size_t)kB * kNA * 16);
    size_t oKeys = alloc((size_t)kB * kG * 8);
    size_t oAcc  = alloc(16);

    float* out     = (float*)d_out;
    float* anchOut = out;
    float* clsOut  = out + kAnchorsN;
    float* regOut  = clsOut + kClsN;
    float* lossOut = regOut + kRegN;

    init_kernel<<<1, 128, 0, stream>>>((unsigned long long*)(ws + oKeys), (float*)(ws + oAcc));

    auto packLaunch = [&](const float* w, size_t off, int MT, int Cout) {
        int total = MT * kCH * 32;
        pack_weights_kernel<<<(total + 255) / 256, 256, 0, stream>>>(
            w, (_Float16*)(ws + off), MT, Cout, total);
    };
    for (int l = 0; l < 4; ++l) {
        packLaunch(regW[l], pRegIn[l], 16, 256);
        packLaunch(clsW[l], pClsIn[l], 16, 256);
    }
    packLaunch(regW[4], pRegF, 3, 36);
    packLaunch(clsW[4], pClsF, 45, 720);

    anchors_kernel<<<(kNA + 255) / 256, 256, 0, stream>>>(anchOut);

    const int Hs[5]   = {80, 40, 20, 10, 5};
    const int AOFF[5] = {0, 57600, 72000, 75600, 76500};
    for (int lev = 0; lev < 5; ++lev) {
        int H = Hs[lev], W = Hs[lev], HW = H * W;
        dim3 tgrid((HW + 31) / 32, 8, kB);
        nchw_to_nhwc_f16_kernel<<<tgrid, 256, 0, stream>>>(
            feat[lev], (_Float16*)(ws + oX), HW);
        dim3 grid64((HW + 63) / 64, kB);
        dim3 grid32((HW + 31) / 32, kB);
        _Float16* bufs[3] = {(_Float16*)(ws + oX), (_Float16*)(ws + oB1), (_Float16*)(ws + oB2)};
        for (int head = 0; head < 2; ++head) {
            const size_t* packs = head ? pClsIn : pRegIn;
            const float* const* biases = head ? clsB : regB;
            int curb = 0, nxt = 1;
            for (int l = 0; l < 4; ++l) {
                conv_wmma_kernel<2, 4, 0><<<grid64, 256, 0, stream>>>(
                    bufs[curb], (const _Float16*)(ws + packs[l]), biases[l],
                    bufs[nxt], nullptr, H, W, 16, 256, 0);
                curb = nxt;
                nxt  = (curb == 1) ? 2 : 1;
            }
            if (head == 0) {
                conv_wmma_kernel<1, 4, 1><<<grid64, 256, 0, stream>>>(
                    bufs[curb], (const _Float16*)(ws + pRegF), regB[4],
                    nullptr, regOut, H, W, 3, 36, AOFF[lev]);
            } else {
                conv_wmma_kernel<6, 2, 2><<<grid32, 256, 0, stream>>>(
                    bufs[curb], (const _Float16*)(ws + pClsF), clsB[4],
                    nullptr, clsOut, H, W, 45, 720, AOFF[lev]);
            }
        }
    }

    dim3 tg((kNA + 255) / 256, kB);
    targets_iou_kernel<<<tg, 256, 0, stream>>>(
        anchOut, gt, ngt, img, (float*)(ws + oMax), (int*)(ws + oArg),
        (unsigned long long*)(ws + oKeys));
    targets_override_kernel<<<1, kB, 0, stream>>>(
        ngt, (const unsigned long long*)(ws + oKeys),
        (float*)(ws + oMax), (int*)(ws + oArg));
    targets_final_kernel<<<tg, 256, 0, stream>>>(
        anchOut, gt, img, (const float*)(ws + oMax), (const int*)(ws + oArg),
        (float*)(ws + oClsT), (float*)(ws + oRegT), (float*)(ws + oAcc));

    loss_cls_kernel<<<(kClsN + 255) / 256, 256, 0, stream>>>(
        clsOut, (const float*)(ws + oClsT), (float*)(ws + oAcc), kClsN);
    loss_reg_kernel<<<(kRegN + 255) / 256, 256, 0, stream>>>(
        regOut, (const float*)(ws + oClsT), (const float*)(ws + oRegT),
        (float*)(ws + oAcc), kRegN);
    finalize_kernel<<<1, 1, 0, stream>>>((const float*)(ws + oAcc), lossOut);
}
